// SparseMLP_63977832841622
// MI455X (gfx1250) — matrix operand order repaired and checked by reading, not compiled, not run on hardware
//
#include <hip/hip_runtime.h>
#include <hip/hip_bf16.h>
#include <math.h>

// ---- CDNA5 WMMA types -------------------------------------------------------
typedef __attribute__((ext_vector_type(16))) __bf16       v16bf;
typedef __attribute__((ext_vector_type(8)))  float        v8f;
typedef __attribute__((ext_vector_type(8)))  unsigned int v8u;

#define NE   64      // experts
#define CAPT 64      // tokens per expert
#define HD   1024    // hidden
#define FD   2048    // ffn hidden
#define KC   64      // K chunk staged in LDS
#define LSTR 72      // padded LDS row stride (ushorts): 144B rows -> conflict-free b128 reads

// Split fp32 into bf16 hi + bf16 lo (truncating split; residual ~2^-16 rel).
__device__ __forceinline__ void split_bf16(float a, unsigned short& h, unsigned short& l) {
    unsigned u = __float_as_uint(a);
    h = (unsigned short)(u >> 16);
    float r = a - __uint_as_float(u & 0xffff0000u);
    l = (unsigned short)(__float_as_uint(r) >> 16);
}

// Build a 16-element bf16 fragment from two 16-byte LDS reads.
__device__ __forceinline__ v16bf ld_frag(const unsigned short* p0, const unsigned short* p1) {
    uint4 a = *(const uint4*)p0;
    uint4 b = *(const uint4*)p1;
    v8u t = {a.x, a.y, a.z, a.w, b.x, b.y, b.z, b.w};
    return __builtin_bit_cast(v16bf, t);
}

// bf16x3 fp32-emulated MAC: acc += (ah+al) * (bh+bl), dropping lo*lo.
__device__ __forceinline__ void mma3(v8f& acc, v16bf ah, v16bf al,
                                     const unsigned short* pbh, const unsigned short* pbl) {
    v16bf bh = ld_frag(pbh, pbh + 8);
    v16bf bl = ld_frag(pbl, pbl + 8);
    acc = __builtin_amdgcn_wmma_f32_16x16x32_bf16(false, ah, false, bh, (short)0, acc, false, false);
    acc = __builtin_amdgcn_wmma_f32_16x16x32_bf16(false, ah, false, bl, (short)0, acc, false, false);
    acc = __builtin_amdgcn_wmma_f32_16x16x32_bf16(false, al, false, bh, (short)0, acc, false, false);
}

__device__ __forceinline__ float gelu_tanh(float v) {
    float c = 0.7978845608028654f * (v + 0.044715f * v * v * v);
    return 0.5f * v * (1.0f + tanhf(c));
}

// ---------------------------------------------------------------------------
// Kernel 1: h[e,m,n] = gelu(sum_k x[e,m,k] * w1[e,n,k])   (sdd + gelu)
// grid.x = NE * (FD/128); block = 256 threads (8 waves), tile M=64 x N=128.
// ---------------------------------------------------------------------------
__global__ __launch_bounds__(256) void moe_gemm1_gelu(const float* __restrict__ x,
                                                      const float* __restrict__ w1,
                                                      float* __restrict__ h) {
    __shared__ unsigned short XH[CAPT * LSTR];
    __shared__ unsigned short XL[CAPT * LSTR];
    __shared__ unsigned short WH[128 * LSTR];
    __shared__ unsigned short WL[128 * LSTR];

    const int tid  = threadIdx.x;
    const int e    = blockIdx.x >> 4;            // FD/128 = 16 N-tiles per expert
    const int n0   = (blockIdx.x & 15) * 128;
    const int lane = tid & 31;
    const int wv   = tid >> 5;                   // 0..7
    const int l15  = lane & 15;
    const int lh   = lane >> 4;                  // lane half
    const int m0   = (wv & 3) * 16;              // wave's M subtile
    const int nb   = (wv >> 2) * 64;             // wave's N strip base

    v8f acc0 = {}, acc1 = {}, acc2 = {}, acc3 = {};

    const float* xrow = x  + (size_t)e * CAPT * HD;
    const float* wrow = w1 + ((size_t)e * FD + n0) * HD;

    for (int k0 = 0; k0 < HD; k0 += KC) {
        __syncthreads();
        // Stage x chunk: 64 rows x 64 K
        for (int i = 0; i < 4; ++i) {
            int idx = tid + 256 * i;
            int m = idx >> 4;
            int c = (idx & 15) << 2;
            float4 v = *(const float4*)(xrow + (size_t)m * HD + k0 + c);
            ushort4 hv, lv;
            split_bf16(v.x, hv.x, lv.x); split_bf16(v.y, hv.y, lv.y);
            split_bf16(v.z, hv.z, lv.z); split_bf16(v.w, hv.w, lv.w);
            *(ushort4*)&XH[m * LSTR + c] = hv;
            *(ushort4*)&XL[m * LSTR + c] = lv;
        }
        // Stage w1 chunk: 128 rows x 64 K  (B[k][n] = w1[n0+n][k], K-contiguous)
        for (int i = 0; i < 8; ++i) {
            int idx = tid + 256 * i;
            int r = idx >> 4;
            int c = (idx & 15) << 2;
            float4 v = *(const float4*)(wrow + (size_t)r * HD + k0 + c);
            ushort4 hv, lv;
            split_bf16(v.x, hv.x, lv.x); split_bf16(v.y, hv.y, lv.y);
            split_bf16(v.z, hv.z, lv.z); split_bf16(v.w, hv.w, lv.w);
            *(ushort4*)&WH[r * LSTR + c] = hv;
            *(ushort4*)&WL[r * LSTR + c] = lv;
        }
        __syncthreads();
        if (k0 + KC < HD)  // warm next weight chunk while we crunch this one
            __builtin_prefetch(wrow + (size_t)(tid >> 1) * HD + k0 + KC + ((tid & 1) << 5), 0, 1);

        for (int kk = 0; kk < KC; kk += 32) {
            const unsigned short* pa  = &XH[(m0 + l15) * LSTR + kk + lh * 16];
            const unsigned short* pal = &XL[(m0 + l15) * LSTR + kk + lh * 16];
            v16bf ah = ld_frag(pa,  pa  + 8);
            v16bf al = ld_frag(pal, pal + 8);
            mma3(acc0, ah, al, &WH[(nb +  0 + l15) * LSTR + kk + lh * 16],
                               &WL[(nb +  0 + l15) * LSTR + kk + lh * 16]);
            mma3(acc1, ah, al, &WH[(nb + 16 + l15) * LSTR + kk + lh * 16],
                               &WL[(nb + 16 + l15) * LSTR + kk + lh * 16]);
            mma3(acc2, ah, al, &WH[(nb + 32 + l15) * LSTR + kk + lh * 16],
                               &WL[(nb + 32 + l15) * LSTR + kk + lh * 16]);
            mma3(acc3, ah, al, &WH[(nb + 48 + l15) * LSTR + kk + lh * 16],
                               &WL[(nb + 48 + l15) * LSTR + kk + lh * 16]);
        }
    }

    // Epilogue: gelu + store to workspace h[E*CAPT, FD]
    float* hout = h + (size_t)e * CAPT * FD + n0;
    for (int i = 0; i < 8; ++i) {
        int m = m0 + lh * 8 + i;
        size_t base = (size_t)m * FD;
        hout[base + nb +  0 + l15] = gelu_tanh(acc0[i]);
        hout[base + nb + 16 + l15] = gelu_tanh(acc1[i]);
        hout[base + nb + 32 + l15] = gelu_tanh(acc2[i]);
        hout[base + nb + 48 + l15] = gelu_tanh(acc3[i]);
    }
}

// ---------------------------------------------------------------------------
// Kernel 2: out[e,m,j] = sum_f h[e,m,f] * w2[e,f,j]       (dsd)
// grid.x = NE * (HD/128); tile M=64 x N=128, K=FD in chunks of 64.
// ---------------------------------------------------------------------------
__global__ __launch_bounds__(256) void moe_gemm2(const float* __restrict__ h,
                                                 const float* __restrict__ w2,
                                                 float* __restrict__ out) {
    __shared__ unsigned short AH[CAPT * LSTR];
    __shared__ unsigned short AL[CAPT * LSTR];
    __shared__ unsigned short BH[128 * LSTR];   // transposed: BH[n][k]
    __shared__ unsigned short BL[128 * LSTR];

    const int tid  = threadIdx.x;
    const int e    = blockIdx.x >> 3;            // HD/128 = 8 N-tiles per expert
    const int j0   = (blockIdx.x & 7) * 128;
    const int lane = tid & 31;
    const int wv   = tid >> 5;
    const int l15  = lane & 15;
    const int lh   = lane >> 4;
    const int m0   = (wv & 3) * 16;
    const int nb   = (wv >> 2) * 64;

    v8f acc0 = {}, acc1 = {}, acc2 = {}, acc3 = {};

    const float* hrow = h  + (size_t)e * CAPT * FD;
    const float* wbase = w2 + (size_t)e * FD * HD + j0;

    for (int k0 = 0; k0 < FD; k0 += KC) {
        __syncthreads();
        // Stage h chunk: 64 rows x 64 K
        for (int i = 0; i < 4; ++i) {
            int idx = tid + 256 * i;
            int m = idx >> 4;
            int c = (idx & 15) << 2;
            float4 v = *(const float4*)(hrow + (size_t)m * FD + k0 + c);
            ushort4 hv, lv;
            split_bf16(v.x, hv.x, lv.x); split_bf16(v.y, hv.y, lv.y);
            split_bf16(v.z, hv.z, lv.z); split_bf16(v.w, hv.w, lv.w);
            *(ushort4*)&AH[m * LSTR + c] = hv;
            *(ushort4*)&AL[m * LSTR + c] = lv;
        }
        // Stage w2 chunk transposed: BH[n][k] = w2[(e*FD+k0+k)*HD + j0+n]
        for (int i = 0; i < 8; ++i) {
            int idx = tid + 256 * i;
            int k = idx >> 5;            // 0..63
            int c = (idx & 31) << 2;     // n 0..124
            float4 v = *(const float4*)(wbase + (size_t)(k0 + k) * HD + c);
            unsigned short hs, ls;
            split_bf16(v.x, hs, ls); BH[(c + 0) * LSTR + k] = hs; BL[(c + 0) * LSTR + k] = ls;
            split_bf16(v.y, hs, ls); BH[(c + 1) * LSTR + k] = hs; BL[(c + 1) * LSTR + k] = ls;
            split_bf16(v.z, hs, ls); BH[(c + 2) * LSTR + k] = hs; BL[(c + 2) * LSTR + k] = ls;
            split_bf16(v.w, hs, ls); BH[(c + 3) * LSTR + k] = hs; BL[(c + 3) * LSTR + k] = ls;
        }
        __syncthreads();
        if (k0 + KC < FD)
            __builtin_prefetch(wbase + (size_t)(k0 + KC + (tid >> 1)) * HD + ((tid & 1) << 6), 0, 1);

        for (int kk = 0; kk < KC; kk += 32) {
            const unsigned short* pa  = &AH[(m0 + l15) * LSTR + kk + lh * 16];
            const unsigned short* pal = &AL[(m0 + l15) * LSTR + kk + lh * 16];
            v16bf ah = ld_frag(pa,  pa  + 8);
            v16bf al = ld_frag(pal, pal + 8);
            mma3(acc0, ah, al, &BH[(nb +  0 + l15) * LSTR + kk + lh * 16],
                               &BL[(nb +  0 + l15) * LSTR + kk + lh * 16]);
            mma3(acc1, ah, al, &BH[(nb + 16 + l15) * LSTR + kk + lh * 16],
                               &BL[(nb + 16 + l15) * LSTR + kk + lh * 16]);
            mma3(acc2, ah, al, &BH[(nb + 32 + l15) * LSTR + kk + lh * 16],
                               &BL[(nb + 32 + l15) * LSTR + kk + lh * 16]);
            mma3(acc3, ah, al, &BH[(nb + 48 + l15) * LSTR + kk + lh * 16],
                               &BL[(nb + 48 + l15) * LSTR + kk + lh * 16]);
        }
    }

    float* obase = out + (size_t)e * CAPT * HD + j0;
    for (int i = 0; i < 8; ++i) {
        int m = m0 + lh * 8 + i;
        size_t base = (size_t)m * HD;
        obase[base + nb +  0 + l15] = acc0[i];
        obase[base + nb + 16 + l15] = acc1[i];
        obase[base + nb + 32 + l15] = acc2[i];
        obase[base + nb + 48 + l15] = acc3[i];
    }
}

extern "C" void kernel_launch(void* const* d_in, const int* in_sizes, int n_in,
                              void* d_out, int out_size, void* d_ws, size_t ws_size,
                              hipStream_t stream) {
    (void)in_sizes; (void)n_in; (void)out_size; (void)ws_size;
    const float* x  = (const float*)d_in[0];
    const float* w1 = (const float*)d_in[1];
    const float* w2 = (const float*)d_in[2];
    float* out  = (float*)d_out;
    float* hbuf = (float*)d_ws;   // needs NE*CAPT*FD*4 = 32 MB scratch

    moe_gemm1_gelu<<<dim3(NE * (FD / 128)), dim3(256), 0, stream>>>(x, w1, hbuf);
    moe_gemm2<<<dim3(NE * (HD / 128)), dim3(256), 0, stream>>>(hbuf, w2, out);
}
